// TransMIL_29549374997069
// MI455X (gfx1250) — matrix operand order repaired and checked by reading, not compiled, not run on hardware
//
#include <hip/hip_runtime.h>
#include <hip/hip_bf16.h>
#include <math.h>

// ---------------- model constants ----------------
constexpr int kHeads = 8;
constexpr int kDh    = 8;
constexpr int kInner = 64;     // kHeads * kDh
constexpr int kLm    = 256;    // landmarks
constexpr int kNtok  = 19600;
constexpr int kN1    = 19601;  // tokens + cls
constexpr int kNp    = 19712;  // padded sequence (=77*256 = 154*128)
constexpr int kPad   = 111;    // front pad
constexpr int kLgrp  = 77;     // tokens per landmark
constexpr int kHid   = 512;
constexpr int kIn    = 1024;
constexpr int kSide  = 140;    // 140*140 == 19600

typedef _Float16 f16;
typedef __attribute__((ext_vector_type(16))) _Float16 v16h;
typedef __attribute__((ext_vector_type(8)))  _Float16 v8h;
typedef __attribute__((ext_vector_type(8)))  float    v8f;
typedef __attribute__((ext_vector_type(2)))  float    v2f;

__device__ inline v8f zero_v8f() {
  v8f v;
  for (int i = 0; i < 8; ++i) v[i] = 0.f;
  return v;
}

// ---------------- casts / copies ----------------
__global__ void k_cast(const float* __restrict__ s, f16* __restrict__ d, long n) {
  long i = (long)blockIdx.x * blockDim.x + threadIdx.x;
  long st = (long)gridDim.x * blockDim.x;
  for (; i < n; i += st) d[i] = (f16)s[i];
}

__global__ void k_copy(const float* __restrict__ s, float* __restrict__ d, int n) {
  int i = blockIdx.x * blockDim.x + threadIdx.x;
  if (i < n) d[i] = s[i];
}

// ---------------- tiled f16 WMMA GEMM ----------------
// C(MxN) = A(MxK,f16) * B(KxN,f16), f32 accumulate, epilogue by mode:
//   mode 0: Out[(m+1)*N+n] = gelu(acc + bias[n])           (fc1 -> h rows 1..)
//   mode 1: Out[m*N+n]     = acc * (n<64 ? 8^-0.5 : 1)     (qkv, fold q scale)
//   mode 2: if (m>=kPad) Out[(m-kPad)*N+n] += acc+bias[n]  (out proj + residual)
// A tile is staged global->LDS with GLOBAL_LOAD_ASYNC_TO_LDS_B128 (ASYNCcnt),
// overlapping with the VALU transpose-staging of the B tile.
__global__ __launch_bounds__(256) void k_gemm(const f16* __restrict__ A,
                                              const f16* __restrict__ B,
                                              const float* __restrict__ bias,
                                              float* __restrict__ Out,
                                              int M, int N, int K, int mode) {
  __shared__ __attribute__((aligned(32))) f16 As[128 * 32];
  __shared__ __attribute__((aligned(32))) f16 Bs[64 * 32];
  const int tid = threadIdx.x;
  const int lane = tid & 31;
  const int wave = tid >> 5;
  const int wm = wave & 3;        // 4 waves along M (32 rows each)
  const int wn = wave >> 2;       // 2 waves along N (32 cols each)
  const int tileM = blockIdx.x * 128;
  const int tileN = blockIdx.y * 64;
  const int lane15 = lane & 15;
  const int laneHi = lane >> 4;

  v8f acc[2][2];
  acc[0][0] = zero_v8f(); acc[0][1] = zero_v8f();
  acc[1][0] = zero_v8f(); acc[1][1] = zero_v8f();

  for (int k0 = 0; k0 < K; k0 += 32) {
    // async-stage A tile: 128 rows x 32 halves (row-major in LDS).
    // Out-of-range rows are clamped to row M-1 (valid memory); their results
    // are discarded by the masked store in the epilogue.
    #pragma unroll
    for (int u = 0; u < 2; ++u) {
      int c = tid + u * 256;            // 512 chunks of 8 halves (16B)
      int row = c >> 2;
      int colh = (c & 3) * 8;
      int gr = tileM + row;
      if (gr >= M) gr = M - 1;
      const f16* gp = A + (size_t)gr * K + k0 + colh;
      unsigned ldsoff = (unsigned)(size_t)(As + row * 32 + colh);
      asm volatile("global_load_async_to_lds_b128 %0, %1, off"
                   :: "v"(ldsoff), "v"((unsigned long long)(size_t)gp)
                   : "memory");
    }
    // stage B tile transposed: Bs[n][k], n in [0,64), k in [0,32)
    #pragma unroll
    for (int u = 0; u < 8; ++u) {
      int e = tid + u * 256;            // 2048 halves
      int kk = e >> 6;
      int nn = e & 63;
      const f16* bp = B + (size_t)(k0 + kk) * N + tileN + nn;
      __builtin_prefetch(bp + 32 * N, 0, 1);   // next k-tile -> global_prefetch_b8
      Bs[nn * 32 + kk] = *bp;
    }
    asm volatile("s_wait_asynccnt 0" ::: "memory");
    __syncthreads();

    #pragma unroll
    for (int tm = 0; tm < 2; ++tm) {
      int rowA = wm * 32 + tm * 16 + lane15;
      const f16* ap = As + rowA * 32 + laneHi * 16;
      v8h alo = *(const v8h*)ap;              // K {0..7}  (+16 for hi lanes)
      v8h ahi = *(const v8h*)(ap + 8);        // K {8..15} (+16 for hi lanes)
      v16h afrag = __builtin_shufflevector(alo, ahi,
          0, 1, 2, 3, 4, 5, 6, 7, 8, 9, 10, 11, 12, 13, 14, 15);
      #pragma unroll
      for (int tn = 0; tn < 2; ++tn) {
        int colB = wn * 32 + tn * 16 + lane15;
        v16h bfrag = *(const v16h*)(Bs + colB * 32 + laneHi * 16);
        acc[tm][tn] = __builtin_amdgcn_wmma_f32_16x16x32_f16(
            false, afrag, false, bfrag, (short)0, acc[tm][tn], false, false);
      }
    }
    __syncthreads();
  }

  const float qscale = 0.35355339059327376f;
  #pragma unroll
  for (int tm = 0; tm < 2; ++tm)
    #pragma unroll
    for (int tn = 0; tn < 2; ++tn)
      #pragma unroll
      for (int v = 0; v < 8; ++v) {
        int m = tileM + wm * 32 + tm * 16 + v + laneHi * 8;
        int n = tileN + wn * 32 + tn * 16 + lane15;
        float val = acc[tm][tn][v];
        if (mode == 0) {
          if (m < M) {
            float xg = val + bias[n];
            Out[(size_t)(m + 1) * N + n] =
                0.5f * xg * (1.0f + erff(xg * 0.70710678118654752f));
          }
        } else if (mode == 1) {
          Out[(size_t)m * N + n] = val * ((n < kInner) ? qscale : 1.0f);
        } else {
          if (m >= kPad) Out[(size_t)(m - kPad) * N + n] += val + bias[n];
        }
      }
}

// ---------------- LayerNorm + front pad -> f16 ----------------
__global__ __launch_bounds__(256) void k_ln_pad(const float* __restrict__ h,
                                                const float* __restrict__ g,
                                                const float* __restrict__ b,
                                                f16* __restrict__ xpn) {
  int i = blockIdx.x;           // 0..kNp-1
  int t = threadIdx.x;
  if (i < kPad) {
    xpn[(size_t)i * kHid + t] = (f16)0;
    xpn[(size_t)i * kHid + t + 256] = (f16)0;
    return;
  }
  const float* row = h + (size_t)(i - kPad) * kHid;
  float x0 = row[t], x1 = row[t + 256];
  __shared__ float red[8];
  __shared__ float stat[2];
  float s = x0 + x1;
  #pragma unroll
  for (int o = 16; o; o >>= 1) s += __shfl_xor(s, o);
  if ((t & 31) == 0) red[t >> 5] = s;
  __syncthreads();
  if (t == 0) { float tt = 0; for (int w = 0; w < 8; ++w) tt += red[w];
                stat[0] = tt * (1.f / kHid); }
  __syncthreads();
  float mu = stat[0];
  float d0 = x0 - mu, d1 = x1 - mu;
  float v = d0 * d0 + d1 * d1;
  #pragma unroll
  for (int o = 16; o; o >>= 1) v += __shfl_xor(v, o);
  __syncthreads();
  if ((t & 31) == 0) red[t >> 5] = v;
  __syncthreads();
  if (t == 0) { float tt = 0; for (int w = 0; w < 8; ++w) tt += red[w];
                stat[1] = rsqrtf(tt * (1.f / kHid) + 1e-5f); }
  __syncthreads();
  float rstd = stat[1];
  xpn[(size_t)i * kHid + t]       = (f16)(d0 * rstd * g[t] + b[t]);
  xpn[(size_t)i * kHid + t + 256] = (f16)(d1 * rstd * g[t + 256] + b[t + 256]);
}

// ---------------- landmark means (q_l, k_l) ----------------
__global__ void k_landmarks(const float* __restrict__ qkv,
                            float* __restrict__ ql, float* __restrict__ kl) {
  int tid = blockIdx.x * blockDim.x + threadIdx.x;  // 2*8*256*8 = 32768
  if (tid >= 2 * kHeads * kLm * kDh) return;
  int d = tid & 7, i = (tid >> 3) & 255, hh = (tid >> 11) & 7, part = tid >> 14;
  int col = part * kInner + hh * kDh + d;
  float s = 0.f;
  for (int j = 0; j < kLgrp; ++j) s += qkv[(size_t)(i * kLgrp + j) * 192 + col];
  float* dst = part ? kl : ql;
  dst[((hh * kLm) + i) * kDh + d] = s * (1.0f / kLgrp);
}

// ---------------- attn2 = softmax(q_l k_l^T) ----------------
__global__ __launch_bounds__(256) void k_attn2(const float* __restrict__ ql,
                                               const float* __restrict__ kl,
                                               float* __restrict__ a) {
  int hh = blockIdx.x >> 8, i = blockIdx.x & 255, j = threadIdx.x;
  const float* q = ql + (hh * kLm + i) * kDh;
  const float* k = kl + (hh * kLm + j) * kDh;
  float l = 0.f;
  #pragma unroll
  for (int d = 0; d < 8; ++d) l += q[d] * k[d];
  __shared__ float red[8];
  __shared__ float stat;
  float m = l;
  #pragma unroll
  for (int o = 16; o; o >>= 1) m = fmaxf(m, __shfl_xor(m, o));
  if ((j & 31) == 0) red[j >> 5] = m;
  __syncthreads();
  if (j == 0) { float mm = red[0]; for (int w = 1; w < 8; ++w) mm = fmaxf(mm, red[w]); stat = mm; }
  __syncthreads();
  float e = __expf(l - stat);
  __syncthreads();
  float s = e;
  #pragma unroll
  for (int o = 16; o; o >>= 1) s += __shfl_xor(s, o);
  if ((j & 31) == 0) red[j >> 5] = s;
  __syncthreads();
  if (j == 0) { float tt = 0; for (int w = 0; w < 8; ++w) tt += red[w]; stat = tt; }
  __syncthreads();
  a[((size_t)(hh * kLm) + i) * kLm + j] = e / stat;
}

// ---------------- per-head max column abs-sum ----------------
__global__ __launch_bounds__(256) void k_colsum(const float* __restrict__ a,
                                                float* __restrict__ colmax) {
  int hh = blockIdx.x, j = threadIdx.x;
  const float* ah = a + (size_t)hh * kLm * kLm;
  float s = 0.f;
  for (int i = 0; i < kLm; ++i) s += fabsf(ah[i * kLm + j]);
  __shared__ float red[8];
  #pragma unroll
  for (int o = 16; o; o >>= 1) s = fmaxf(s, __shfl_xor(s, o));
  if ((j & 31) == 0) red[j >> 5] = s;
  __syncthreads();
  if (j == 0) { float m = red[0]; for (int w = 1; w < 8; ++w) m = fmaxf(m, red[w]); colmax[hh] = m; }
}

// ---------------- z0 = a^T / max_global_colsum ----------------
__global__ void k_initz(const float* __restrict__ a,
                        const float* __restrict__ colmax, float* __restrict__ z) {
  int tid = blockIdx.x * 256 + threadIdx.x;   // 8*256*256
  int j = tid & 255, i = (tid >> 8) & 255, hh = tid >> 16;
  float cm = colmax[0];
  #pragma unroll
  for (int w = 1; w < 8; ++w) cm = fmaxf(cm, colmax[w]);
  z[((size_t)hh * kLm + i) * kLm + j] = a[((size_t)hh * kLm + j) * kLm + i] / cm;
}

// ---------------- batched 256x256 matmul, C = alpha*(A@B) + beta*I --------
// grid.x = 8 heads * 16 (64x64 tiles); 8 waves; each wave = 16x32 via f32 WMMA
__global__ __launch_bounds__(256) void k_mm256(float* __restrict__ C,
                                               const float* __restrict__ A,
                                               const float* __restrict__ B,
                                               float alpha, float beta) {
  int bid = blockIdx.x;
  int hh = bid >> 4, t = bid & 15;
  int tileM = (t >> 2) * 64, tileN = (t & 3) * 64;
  const float* Ah = A + (size_t)hh * kLm * kLm;
  const float* Bh = B + (size_t)hh * kLm * kLm;
  float* Ch = C + (size_t)hh * kLm * kLm;
  int lane = threadIdx.x & 31, wave = threadIdx.x >> 5;
  int m0 = tileM + (wave & 3) * 16;
  int n0 = tileN + (wave >> 2) * 32;
  int lane15 = lane & 15, kb = (lane >> 4) * 2;
#if __has_builtin(__builtin_amdgcn_wmma_f32_16x16x4_f32)
  v8f acc0 = zero_v8f(), acc1 = zero_v8f();
  const float* arow = Ah + (size_t)(m0 + lane15) * kLm;
  for (int k0 = 0; k0 < kLm; k0 += 4) {
    v2f a; a[0] = arow[k0 + kb]; a[1] = arow[k0 + kb + 1];
    v2f b0; b0[0] = Bh[(size_t)(k0 + kb) * kLm + n0 + lane15];
            b0[1] = Bh[(size_t)(k0 + kb + 1) * kLm + n0 + lane15];
    v2f b1; b1[0] = Bh[(size_t)(k0 + kb) * kLm + n0 + 16 + lane15];
            b1[1] = Bh[(size_t)(k0 + kb + 1) * kLm + n0 + 16 + lane15];
    acc0 = __builtin_amdgcn_wmma_f32_16x16x4_f32(false, a, false, b0, (short)0, acc0, false, false);
    acc1 = __builtin_amdgcn_wmma_f32_16x16x4_f32(false, a, false, b1, (short)0, acc1, false, false);
  }
  #pragma unroll
  for (int v = 0; v < 8; ++v) {
    int m = m0 + v + (lane >> 4) * 8;
    int na = n0 + lane15, nb = n0 + 16 + lane15;
    Ch[(size_t)m * kLm + na] = alpha * acc0[v] + ((m == na) ? beta : 0.f);
    Ch[(size_t)m * kLm + nb] = alpha * acc1[v] + ((m == nb) ? beta : 0.f);
  }
#else
  for (int idx = lane; idx < 16 * 32; idx += 32) {
    int mm = m0 + (idx & 15), nn = n0 + (idx >> 4);
    float s = 0.f;
    for (int k = 0; k < kLm; ++k) s += Ah[(size_t)mm * kLm + k] * Bh[(size_t)k * kLm + nn];
    Ch[(size_t)mm * kLm + nn] = alpha * s + ((mm == nn) ? beta : 0.f);
  }
#endif
}

// ---------------- y = beta*I - x (elementwise) ----------------
__global__ void k_axmb(float* __restrict__ y, const float* __restrict__ x, float beta) {
  int tid = blockIdx.x * 256 + threadIdx.x;   // 8*256*256
  int j = tid & 255, i = (tid >> 8) & 255;
  y[tid] = ((i == j) ? beta : 0.f) - x[tid];
}

// ---------------- av = softmax(q_l k^T) @ v (online, one wave per (h,i)) ----
__global__ __launch_bounds__(256) void k_attn3v(const float* __restrict__ qkv,
                                                const float* __restrict__ ql,
                                                float* __restrict__ av) {
  int g = blockIdx.x * 8 + (threadIdx.x >> 5);    // 0..2047
  int lane = threadIdx.x & 31;
  int hh = g >> 8, i = g & 255;
  const float* qp = ql + (hh * kLm + i) * kDh;
  float q[8];
  #pragma unroll
  for (int d = 0; d < 8; ++d) q[d] = qp[d];
  float m = -1e30f, s = 0.f, acc[8] = {};
  for (int j = lane; j < kNp; j += 32) {
    const float* kr = qkv + (size_t)j * 192 + kInner + hh * kDh;
    const float* vr = kr + kInner;
    float l = 0.f;
    #pragma unroll
    for (int d = 0; d < 8; ++d) l += q[d] * kr[d];
    float nm = fmaxf(m, l);
    float corr = __expf(m - nm);
    float p = __expf(l - nm);
    s = s * corr + p;
    #pragma unroll
    for (int d = 0; d < 8; ++d) acc[d] = acc[d] * corr + p * vr[d];
    m = nm;
  }
  #pragma unroll
  for (int o = 16; o; o >>= 1) {
    float m2 = __shfl_xor(m, o), s2 = __shfl_xor(s, o);
    float nm = fmaxf(m, m2);
    float c1 = __expf(m - nm), c2 = __expf(m2 - nm);
    s = s * c1 + s2 * c2;
    #pragma unroll
    for (int d = 0; d < 8; ++d) {
      float a2 = __shfl_xor(acc[d], o);
      acc[d] = acc[d] * c1 + a2 * c2;
    }
    m = nm;
  }
  float outv = 0.f;
  #pragma unroll
  for (int d = 0; d < 8; ++d) if (lane == d) outv = acc[d];
  if (lane < 8) av[((size_t)hh * kLm + i) * kDh + lane] = outv / s;
}

// ---------------- w2 = pinv(attn2) @ av ----------------
__global__ void k_zav(const float* __restrict__ z, const float* __restrict__ av,
                      float* __restrict__ w2) {
  int tid = blockIdx.x * 256 + threadIdx.x;    // 8*256*8
  if (tid >= kHeads * kLm * kDh) return;
  int d = tid & 7, i = (tid >> 3) & 255, hh = tid >> 11;
  const float* zr = z + ((size_t)hh * kLm + i) * kLm;
  const float* avh = av + (size_t)hh * kLm * kDh;
  float s = 0.f;
  for (int j = 0; j < kLm; ++j) s += zr[j] * avh[j * kDh + d];
  w2[((size_t)hh * kLm + i) * kDh + d] = s;
}

// ---------------- out = softmax(q k_l^T) @ w2 + dwconv33(v); one wave/(h,i) -
__global__ __launch_bounds__(256) void k_attn1out(const float* __restrict__ qkv,
                                                  const float* __restrict__ kl,
                                                  const float* __restrict__ w2,
                                                  const float* __restrict__ rw,
                                                  f16* __restrict__ obuf) {
  int i = blockIdx.x;                 // token 0..kNp-1
  int hh = threadIdx.x >> 5;          // wave = head
  int lane = threadIdx.x & 31;
  const float* qp = qkv + (size_t)i * 192 + hh * kDh;
  float q[8];
  #pragma unroll
  for (int d = 0; d < 8; ++d) q[d] = qp[d];
  float l[8], lm = -1e30f;
  #pragma unroll
  for (int u = 0; u < 8; ++u) {
    int j = lane * 8 + u;
    const float* kr = kl + ((size_t)hh * kLm + j) * kDh;
    float t = 0.f;
    #pragma unroll
    for (int d = 0; d < 8; ++d) t += q[d] * kr[d];
    l[u] = t; lm = fmaxf(lm, t);
  }
  #pragma unroll
  for (int o = 16; o; o >>= 1) lm = fmaxf(lm, __shfl_xor(lm, o));
  float s = 0.f, acc[8] = {};
  #pragma unroll
  for (int u = 0; u < 8; ++u) {
    int j = lane * 8 + u;
    float e = __expf(l[u] - lm);
    s += e;
    const float* wr = w2 + ((size_t)hh * kLm + j) * kDh;
    #pragma unroll
    for (int d = 0; d < 8; ++d) acc[d] += e * wr[d];
  }
  #pragma unroll
  for (int o = 16; o; o >>= 1) {
    s += __shfl_xor(s, o);
    #pragma unroll
    for (int d = 0; d < 8; ++d) acc[d] += __shfl_xor(acc[d], o);
  }
  if (lane < 8) {
    int d = lane;
    float conv = 0.f;
    const float* rwh = rw + hh * 33;
    #pragma unroll
    for (int tt = 0; tt < 33; ++tt) {
      int r = i + tt - 16;
      if (r >= 0 && r < kNp)
        conv += rwh[tt] * qkv[(size_t)r * 192 + 2 * kInner + hh * kDh + d];
    }
    float accd = 0.f;
    #pragma unroll
    for (int u = 0; u < 8; ++u) if (d == u) accd = acc[u];
    obuf[(size_t)i * kInner + hh * kDh + d] = (f16)(accd / s + conv);
  }
}

// ---------------- PPEG: fused depthwise 7x7 + 5x5 + 3x3 + identity ---------
__global__ void k_ppeg(const float* __restrict__ hin,
                       const float* __restrict__ w7, const float* __restrict__ b7,
                       const float* __restrict__ w5, const float* __restrict__ b5,
                       const float* __restrict__ w3, const float* __restrict__ b3,
                       float* __restrict__ hout) {
  int tid = blockIdx.x * 256 + threadIdx.x;
  if (tid >= kNtok * kHid) return;
  int c = tid & (kHid - 1);
  int pix = tid >> 9;
  int y = pix / kSide, x = pix % kSide;
  const float* f = hin + kHid;                   // skip cls row
  float acc = f[(size_t)pix * kHid + c] + b7[c] + b5[c] + b3[c];
  const float* wc7 = w7 + c * 49;
  #pragma unroll
  for (int dy = -3; dy <= 3; ++dy) {
    int yy = y + dy; if ((unsigned)yy >= (unsigned)kSide) continue;
    #pragma unroll
    for (int dx = -3; dx <= 3; ++dx) {
      int xx = x + dx; if ((unsigned)xx >= (unsigned)kSide) continue;
      acc += wc7[(dy + 3) * 7 + (dx + 3)] * f[(size_t)(yy * kSide + xx) * kHid + c];
    }
  }
  const float* wc5 = w5 + c * 25;
  #pragma unroll
  for (int dy = -2; dy <= 2; ++dy) {
    int yy = y + dy; if ((unsigned)yy >= (unsigned)kSide) continue;
    #pragma unroll
    for (int dx = -2; dx <= 2; ++dx) {
      int xx = x + dx; if ((unsigned)xx >= (unsigned)kSide) continue;
      acc += wc5[(dy + 2) * 5 + (dx + 2)] * f[(size_t)(yy * kSide + xx) * kHid + c];
    }
  }
  const float* wc3 = w3 + c * 9;
  #pragma unroll
  for (int dy = -1; dy <= 1; ++dy) {
    int yy = y + dy; if ((unsigned)yy >= (unsigned)kSide) continue;
    #pragma unroll
    for (int dx = -1; dx <= 1; ++dx) {
      int xx = x + dx; if ((unsigned)xx >= (unsigned)kSide) continue;
      acc += wc3[(dy + 1) * 3 + (dx + 1)] * f[(size_t)(yy * kSide + xx) * kHid + c];
    }
  }
  hout[kHid + (size_t)pix * kHid + c] = acc;
}

// ---------------- final LayerNorm on row 0 -> d_out ----------------
__global__ __launch_bounds__(256) void k_final_ln(const float* __restrict__ h,
                                                  const float* __restrict__ g,
                                                  const float* __restrict__ b,
                                                  float* __restrict__ out) {
  int t = threadIdx.x;
  float x0 = h[t], x1 = h[t + 256];
  __shared__ float red[8];
  __shared__ float stat[2];
  float s = x0 + x1;
  #pragma unroll
  for (int o = 16; o; o >>= 1) s += __shfl_xor(s, o);
  if ((t & 31) == 0) red[t >> 5] = s;
  __syncthreads();
  if (t == 0) { float tt = 0; for (int w = 0; w < 8; ++w) tt += red[w];
                stat[0] = tt * (1.f / kHid); }
  __syncthreads();
  float mu = stat[0];
  float d0 = x0 - mu, d1 = x1 - mu;
  float v = d0 * d0 + d1 * d1;
  #pragma unroll
  for (int o = 16; o; o >>= 1) v += __shfl_xor(v, o);
  __syncthreads();
  if ((t & 31) == 0) red[t >> 5] = v;
  __syncthreads();
  if (t == 0) { float tt = 0; for (int w = 0; w < 8; ++w) tt += red[w];
                stat[1] = rsqrtf(tt * (1.f / kHid) + 1e-5f); }
  __syncthreads();
  float rstd = stat[1];
  out[t]       = d0 * rstd * g[t] + b[t];
  out[t + 256] = d1 * rstd * g[t + 256] + b[t + 256];
}

// ---------------- host-side orchestration ----------------
static inline size_t al256(size_t v) { return (v + 255) & ~(size_t)255; }

struct Ws {
  f16*   xpn;  float* qkvb; f16* obuf;
  float *h1, *h2;
  f16   *w1h, *qkvw1, *qkvw2, *outw1, *outw2;
  float *amat, *azm, *ya, *yb, *za, *zb;
  float *ql, *kl, *av, *w2m, *colmax;
  f16*   xh;
};

static void run_nystrom(float* h, const float* g, const float* b,
                        const f16* qkvw, const f16* outw, const float* outb,
                        const float* resw, const Ws& W, hipStream_t stream) {
  k_ln_pad<<<kNp, 256, 0, stream>>>(h, g, b, W.xpn);
  dim3 gq(kNp / 128, 192 / 64);
  k_gemm<<<gq, 256, 0, stream>>>(W.xpn, qkvw, nullptr, W.qkvb, kNp, 192, kHid, 1);
  k_landmarks<<<128, 256, 0, stream>>>(W.qkvb, W.ql, W.kl);
  k_attn2<<<kHeads * kLm, 256, 0, stream>>>(W.ql, W.kl, W.amat);
  k_colsum<<<kHeads, 256, 0, stream>>>(W.amat, W.colmax);
  k_initz<<<kHeads * kLm, 256, 0, stream>>>(W.amat, W.colmax, W.za);
  float* zc = W.za;
  float* zn = W.zb;
  for (int it = 0; it < 6; ++it) {
    k_mm256<<<128, 256, 0, stream>>>(W.azm, W.amat, zc, 1.f, 0.f);      // az = a@z
    k_axmb<<<kHeads * kLm, 256, 0, stream>>>(W.ya, W.azm, 7.f);         // 7I - az
    k_mm256<<<128, 256, 0, stream>>>(W.yb, W.azm, W.ya, -1.f, 15.f);    // 15I - az@(.)
    k_mm256<<<128, 256, 0, stream>>>(W.ya, W.azm, W.yb, -1.f, 13.f);    // 13I - az@(.)
    k_mm256<<<128, 256, 0, stream>>>(zn, zc, W.ya, 0.25f, 0.f);         // z' = .25 z@(.)
    float* tmp = zc; zc = zn; zn = tmp;
  }
  k_attn3v<<<kHeads * kLm / 8, 256, 0, stream>>>(W.qkvb, W.ql, W.av);
  k_zav<<<kHeads * kLm * kDh / 256, 256, 0, stream>>>(zc, W.av, W.w2m);
  k_attn1out<<<kNp, 256, 0, stream>>>(W.qkvb, W.kl, W.w2m, resw, W.obuf);
  dim3 go(kNp / 128, kHid / 64);
  k_gemm<<<go, 256, 0, stream>>>(W.obuf, outw, outb, h, kNp, kHid, kInner, 2);
}

extern "C" void kernel_launch(void* const* d_in, const int* in_sizes, int n_in,
                              void* d_out, int out_size, void* d_ws, size_t ws_size,
                              hipStream_t stream) {
  const float* x       = (const float*)d_in[0];
  const float* fc1_w   = (const float*)d_in[1];
  const float* fc1_b   = (const float*)d_in[2];
  const float* cls_tok = (const float*)d_in[3];
  const float* l1_g    = (const float*)d_in[4];
  const float* l1_b    = (const float*)d_in[5];
  const float* l1_qkvw = (const float*)d_in[6];
  const float* l1_outw = (const float*)d_in[7];
  const float* l1_outb = (const float*)d_in[8];
  const float* l1_resw = (const float*)d_in[9];
  const float* l2_g    = (const float*)d_in[10];
  const float* l2_b    = (const float*)d_in[11];
  const float* l2_qkvw = (const float*)d_in[12];
  const float* l2_outw = (const float*)d_in[13];
  const float* l2_outb = (const float*)d_in[14];
  const float* l2_resw = (const float*)d_in[15];
  const float* p7w = (const float*)d_in[16];
  const float* p7b = (const float*)d_in[17];
  const float* p5w = (const float*)d_in[18];
  const float* p5b = (const float*)d_in[19];
  const float* p3w = (const float*)d_in[20];
  const float* p3b = (const float*)d_in[21];
  const float* ng  = (const float*)d_in[22];
  const float* nb  = (const float*)d_in[23];

  // ---- workspace layout (region 0 is reused: xh first, then xpn/qkv/obuf) --
  char* base = (char*)d_ws;
  const size_t XH_B   = (size_t)kNtok * kIn * sizeof(f16);
  const size_t XPN_B  = al256((size_t)kNp * kHid * sizeof(f16));
  const size_t QKV_B  = al256((size_t)kNp * 192 * sizeof(float));
  const size_t OBUF_B = al256((size_t)kNp * kInner * sizeof(f16));
  const size_t R0 = al256((XH_B > XPN_B + QKV_B + OBUF_B) ? XH_B
                                                          : XPN_B + QKV_B + OBUF_B);
  Ws W;
  W.xh   = (f16*)base;
  W.xpn  = (f16*)base;
  W.qkvb = (float*)(base + XPN_B);
  W.obuf = (f16*)(base + XPN_B + QKV_B);
  size_t off = R0;
  auto take = [&](size_t bytes) -> char* { char* p = base + off; off += al256(bytes); return p; };
  W.h1    = (float*)take((size_t)kN1 * kHid * 4);
  W.h2    = (float*)take((size_t)kN1 * kHid * 4);
  W.w1h   = (f16*)take((size_t)kIn * kHid * 2);
  W.qkvw1 = (f16*)take((size_t)kHid * 192 * 2);
  W.qkvw2 = (f16*)take((size_t)kHid * 192 * 2);
  W.outw1 = (f16*)take((size_t)kInner * kHid * 2);
  W.outw2 = (f16*)take((size_t)kInner * kHid * 2);
  W.amat  = (float*)take((size_t)kHeads * kLm * kLm * 4);
  W.azm   = (float*)take((size_t)kHeads * kLm * kLm * 4);
  W.ya    = (float*)take((size_t)kHeads * kLm * kLm * 4);
  W.yb    = (float*)take((size_t)kHeads * kLm * kLm * 4);
  W.za    = (float*)take((size_t)kHeads * kLm * kLm * 4);
  W.zb    = (float*)take((size_t)kHeads * kLm * kLm * 4);
  W.ql    = (float*)take((size_t)kHeads * kLm * kDh * 4);
  W.kl    = (float*)take((size_t)kHeads * kLm * kDh * 4);
  W.av    = (float*)take((size_t)kHeads * kLm * kDh * 4);
  W.w2m   = (float*)take((size_t)kHeads * kLm * kDh * 4);
  W.colmax = (float*)take(256);

  // ---- f16 staging of activations / weights ----
  k_cast<<<1024, 256, 0, stream>>>(x, W.xh, (long)kNtok * kIn);
  k_cast<<<512, 256, 0, stream>>>(fc1_w, W.w1h, (long)kIn * kHid);
  k_cast<<<96, 256, 0, stream>>>(l1_qkvw, W.qkvw1, (long)kHid * 192);
  k_cast<<<96, 256, 0, stream>>>(l2_qkvw, W.qkvw2, (long)kHid * 192);
  k_cast<<<32, 256, 0, stream>>>(l1_outw, W.outw1, (long)kInner * kHid);
  k_cast<<<32, 256, 0, stream>>>(l2_outw, W.outw2, (long)kInner * kHid);

  // ---- fc1 + GELU -> h1 rows 1..19600 ; cls -> row 0 ----
  dim3 g1((kNtok + 127) / 128, kHid / 64);
  k_gemm<<<g1, 256, 0, stream>>>(W.xh, W.w1h, fc1_b, W.h1, kNtok, kHid, kIn, 0);
  k_copy<<<2, 256, 0, stream>>>(cls_tok, W.h1, kHid);

  // ---- layer 1 ----
  run_nystrom(W.h1, l1_g, l1_b, W.qkvw1, W.outw1, l1_outb, l1_resw, W, stream);

  // ---- PPEG h1 -> h2 ----
  k_ppeg<<<(kNtok * kHid) / 256, 256, 0, stream>>>(W.h1, p7w, p7b, p5w, p5b, p3w, p3b, W.h2);
  k_copy<<<2, 256, 0, stream>>>(W.h1, W.h2, kHid);   // cls row passthrough

  // ---- layer 2 ----
  run_nystrom(W.h2, l2_g, l2_b, W.qkvw2, W.outw2, l2_outb, l2_resw, W, stream);

  // ---- final LN of row 0 -> d_out (512 f32) ----
  k_final_ln<<<1, 256, 0, stream>>>(W.h2, ng, nb, (float*)d_out);
}
